// edge_sim_conv_82179904242036
// MI455X (gfx1250) — compile-verified, matrix operand order checked by reading, not run on hardware
//
#include <hip/hip_runtime.h>
#include <hip/hip_bf16.h>

typedef __attribute__((ext_vector_type(16))) _Float16 v16h;
typedef __attribute__((ext_vector_type(8)))  _Float16 h8;
typedef __attribute__((ext_vector_type(8)))  float    v8f;

constexpr int Bq = 2, Nq = 8192, Mq = 4096, Cq = 64, Kq = 16, HIDq = 256;
constexpr float EPS_BN = 1e-5f;

// half index -> K offset within a 32-deep f16 WMMA fragment (A and B share striping)
__device__ __forceinline__ int kmap_half(int lane, int h) {
    return ((lane < 16) ? 0 : 8) + h + ((h < 8) ? 0 : 8);
}

// ---------------------------------------------------------------------------
// Kernel 1: nearest clean point (argmin) + top-16 KNN of that point, LDS-tiled
// ---------------------------------------------------------------------------
constexpr int TILE = 512;

__global__ __launch_bounds__(256) void knn_kernel(const float* __restrict__ pcl,
                                                  const float* __restrict__ noise,
                                                  int* __restrict__ close_idx,
                                                  int* __restrict__ knn)
{
    __shared__ float tp[TILE * 3];
    const int tid = threadIdx.x;
    const int gid = blockIdx.x * 256 + tid;            // one noisy point per thread
    const int b   = gid / Nq;                          // uniform within a block
    const float qx = noise[gid * 3 + 0];
    const float qy = noise[gid * 3 + 1];
    const float qz = noise[gid * 3 + 2];
    const float* pb = pcl + (size_t)b * Mq * 3;

    // Pass A: argmin distance(noise, pcl)
    float bestd = 3.4e38f; int besti = 0;
    for (int t0 = 0; t0 < Mq; t0 += TILE) {
        __syncthreads();
        for (int i = tid; i < TILE * 3; i += 256) tp[i] = pb[t0 * 3 + i];
        __syncthreads();
        for (int j = 0; j < TILE; ++j) {
            float dx = tp[j*3+0] - qx, dy = tp[j*3+1] - qy, dz = tp[j*3+2] - qz;
            float d = dx*dx + dy*dy + dz*dz;
            if (d < bestd) { bestd = d; besti = t0 + j; }
        }
    }
    close_idx[gid] = besti;
    const float cx = pb[besti*3+0], cy = pb[besti*3+1], cz = pb[besti*3+2];

    // Pass B: top-16 smallest distance(close_point, pcl); index-stable ties
    float ad[Kq]; int ai[Kq];
    for (int i = 0; i < Kq; ++i) { ad[i] = 3.4e38f; ai[i] = 0; }
    for (int t0 = 0; t0 < Mq; t0 += TILE) {
        __syncthreads();
        for (int i = tid; i < TILE * 3; i += 256) tp[i] = pb[t0 * 3 + i];
        __syncthreads();
        for (int j = 0; j < TILE; ++j) {
            float dx = tp[j*3+0] - cx, dy = tp[j*3+1] - cy, dz = tp[j*3+2] - cz;
            float d = dx*dx + dy*dy + dz*dz;
            if (d < ad[Kq - 1]) {
                int pos = Kq - 1;
                while (pos > 0 && d < ad[pos - 1]) {
                    ad[pos] = ad[pos - 1]; ai[pos] = ai[pos - 1]; --pos;
                }
                ad[pos] = d; ai[pos] = t0 + j;
            }
        }
    }
    // drop self (element 0), keep 15 neighbors
    for (int i = 1; i < Kq; ++i) knn[(size_t)gid * (Kq - 1) + (i - 1)] = ai[i];
}

// ---------------------------------------------------------------------------
// Kernel 2: per-point weights + WMMA MLP 6->256->64, weighted sums, df rows
// One wave == one noisy point (15 neighbors + 1 pad row == one 16-row tile).
// ---------------------------------------------------------------------------
constexpr int W2 = 2;   // waves per block

__global__ __launch_bounds__(64) void point_mlp_kernel(
    const float* __restrict__ pcl, const float* __restrict__ noise,
    const float* __restrict__ feature,
    const float* __restrict__ w2a, const float* __restrict__ b2a,
    const float* __restrict__ g2a, const float* __restrict__ bt2a,
    const float* __restrict__ w2b, const float* __restrict__ b2b,
    const int* __restrict__ close_idx, const int* __restrict__ knn,
    _Float16* __restrict__ df)
{
    __shared__ __attribute__((aligned(32))) _Float16 b2f[32 * 32 * 16]; // 32 KB: [frag=kc*4+nt][lane][16]
    __shared__ float s2[HIDq], t2[HIDq], bb2[Cq];
    __shared__ __attribute__((aligned(16))) _Float16 hstage[W2][16][HIDq]; // 16 KB

    const int tid  = threadIdx.x;
    const int lane = tid & 31;
    const int wave = tid >> 5;

    // Fill BN params and layer-2 B fragments (swizzled so each lane reads 32B contiguous)
    for (int c = tid; c < HIDq; c += 64) {
        float s = g2a[c] * rsqrtf(1.f + EPS_BN);
        s2[c] = s; t2[c] = b2a[c] * s + bt2a[c];
    }
    for (int c = tid; c < Cq; c += 64) bb2[c] = b2b[c];
    for (int i = tid; i < 32 * 32 * 16; i += 64) {
        int h = i & 15, l = (i >> 4) & 31, f = i >> 9;
        int kc = f >> 2, nt = f & 3;
        int k = 32 * kc + kmap_half(l, h);
        int col = nt * 16 + (l & 15);
        b2f[i] = (_Float16)w2b[col * HIDq + k];
    }

    // Layer-1 B fragments live in registers (K=6 padded to 32, N=256 -> 16 frags)
    v16h b1f[16];
    #pragma unroll
    for (int nt = 0; nt < 16; ++nt) {
        #pragma unroll
        for (int h = 0; h < 16; ++h) {
            int k = kmap_half(lane, h);
            int col = nt * 16 + (lane & 15);
            b1f[nt][h] = (k < 6) ? (_Float16)w2a[col * 6 + k] : (_Float16)0.f;
        }
    }
    __syncthreads();

    const int nwaves = gridDim.x * W2;          // 512 -> exactly 32 points per wave
    const int m16 = lane & 15;
    const int hi8 = (lane >= 16) ? 8 : 0;

    for (int p = blockIdx.x * W2 + wave; p < Bq * Nq; p += nwaves) {
        const int b = p / Nq;
        const float* pb = pcl + (size_t)b * Mq * 3;
        const float* fb = feature + (size_t)b * Mq * Cq;
        const int ci = close_idx[p];
        const float nx = noise[p*3+0], ny = noise[p*3+1], nz = noise[p*3+2];
        const float cpx = pb[ci*3+0], cpy = pb[ci*3+1], cpz = pb[ci*3+2];

        int myidx = 0;
        float dx = 0.f, dy = 0.f, dz = 0.f, e = 0.f;
        if (lane < Kq - 1) {                      // lanes 0..14 own neighbor rows
            myidx = knn[(size_t)p * (Kq - 1) + lane];
            float px = pb[myidx*3+0], py = pb[myidx*3+1], pz = pb[myidx*3+2];
            dx = px - nx; dy = py - ny; dz = pz - nz;
            e = __expf(-10.f * sqrtf(dx*dx + dy*dy + dz*dz));
        }
        float se = e;
        #pragma unroll
        for (int off = 16; off; off >>= 1) se += __shfl_xor(se, off);
        const float wgt = e / (se + 1e-7f);
        float wm[8];
        #pragma unroll
        for (int r = 0; r < 8; ++r) wm[r] = __shfl(wgt, hi8 + r);

        // Layer-1 A fragment: row m=lane (<15), K=0..5 = [delta, close_point]
        v16h a1;
        #pragma unroll
        for (int h = 0; h < 16; ++h) a1[h] = (_Float16)0.f;
        if (lane < Kq - 1) {
            a1[0] = (_Float16)dx;  a1[1] = (_Float16)dy;  a1[2] = (_Float16)dz;
            a1[3] = (_Float16)cpx; a1[4] = (_Float16)cpy; a1[5] = (_Float16)cpz;
        }

        // Layer 1: 16 WMMAs -> hidden 16x256, BN+ReLU, stash f16 in LDS (transpose)
        #pragma unroll
        for (int nt = 0; nt < 16; ++nt) {
            v8f c = {};
            c = __builtin_amdgcn_wmma_f32_16x16x32_f16(false, a1, false, b1f[nt],
                                                       (short)0, c, false, false);
            const int col = nt * 16 + m16;
            const float s = s2[col], t = t2[col];
            #pragma unroll
            for (int r = 0; r < 8; ++r)
                hstage[wave][hi8 + r][col] = (_Float16)fmaxf(c[r] * s + t, 0.f);
        }
        __syncthreads();

        // Layer-2 A fragments from LDS transpose (two 16B runs each)
        const h8* hrow = (const h8*)&hstage[wave][m16][0];
        v16h a2[8];
        #pragma unroll
        for (int kc = 0; kc < 8; ++kc) {
            h8 lo = hrow[4*kc + ((lane < 16) ? 0 : 1)];
            h8 hi = hrow[4*kc + ((lane < 16) ? 2 : 3)];
            a2[kc] = __builtin_shufflevector(lo, hi, 0,1,2,3,4,5,6,7,8,9,10,11,12,13,14,15);
        }

        // Layer 2: 32 WMMAs -> pf 16x64; weighted row-reduce -> point_feature
        #pragma unroll
        for (int nt = 0; nt < 4; ++nt) {
            v8f c = {};
            #pragma unroll
            for (int kc = 0; kc < 8; ++kc) {
                const v16h bf = *(const v16h*)&b2f[((kc * 4 + nt) * 32 + lane) * 16];
                c = __builtin_amdgcn_wmma_f32_16x16x32_f16(false, a2[kc], false, bf,
                                                           (short)0, c, false, false);
            }
            const int col = nt * 16 + m16;
            const float bias = bb2[col];
            float part = 0.f;
            #pragma unroll
            for (int r = 0; r < 8; ++r) part += wm[r] * (c[r] + bias);
            part += __shfl_xor(part, 16);       // combine M=0..7 and M=8..15 halves
            if (lane < 16) df[(size_t)p * 192 + 128 + col] = (_Float16)part;
        }

        // close_feat and co_feature (2 channels per lane)
        #pragma unroll
        for (int q = 0; q < 2; ++q) {
            const int cc = lane + 32 * q;
            float cf = fb[(size_t)ci * Cq + cc];
            float co = 0.f;
            for (int k = 0; k < Kq - 1; ++k) {
                const float wk = __shfl(wgt, k);
                const int   ik = __shfl(myidx, k);
                co += wk * fb[(size_t)ik * Cq + cc];
            }
            df[(size_t)p * 192 + cc]      = (_Float16)cf;
            df[(size_t)p * 192 + 64 + cc] = (_Float16)co;
        }
        __syncthreads();
    }
}

// ---------------------------------------------------------------------------
// Kernel 3: final MLP 192->64 (BN+ReLU) ->64 over 16-row tiles, WMMA
// ---------------------------------------------------------------------------
constexpr int W3 = 8;

__global__ __launch_bounds__(256) void final_mlp_kernel(
    const _Float16* __restrict__ df,
    const float* __restrict__ w1a, const float* __restrict__ b1a,
    const float* __restrict__ g1a, const float* __restrict__ bt1a,
    const float* __restrict__ w1b, const float* __restrict__ b1b,
    float* __restrict__ out)
{
    __shared__ __attribute__((aligned(32))) _Float16 bfa[24 * 32 * 16]; // 24 KB
    __shared__ __attribute__((aligned(32))) _Float16 bfb[8 * 32 * 16];  //  8 KB
    __shared__ float s1[Cq], t1[Cq], bb1[Cq];
    __shared__ __attribute__((aligned(16))) _Float16 h1s[W3][16][Cq];   // 16 KB

    const int tid = threadIdx.x, lane = tid & 31, wave = tid >> 5;

    for (int c = tid; c < Cq; c += 256) {
        float s = g1a[c] * rsqrtf(1.f + EPS_BN);
        s1[c] = s; t1[c] = b1a[c] * s + bt1a[c]; bb1[c] = b1b[c];
    }
    for (int i = tid; i < 24 * 32 * 16; i += 256) {
        int h = i & 15, l = (i >> 4) & 31, f = i >> 9;
        int kc = f >> 2, nt = f & 3;
        int k = 32 * kc + kmap_half(l, h);
        int col = nt * 16 + (l & 15);
        bfa[i] = (_Float16)w1a[col * 192 + k];
    }
    for (int i = tid; i < 8 * 32 * 16; i += 256) {
        int h = i & 15, l = (i >> 4) & 31, f = i >> 9;
        int kc = f >> 2, nt = f & 3;
        int k = 32 * kc + kmap_half(l, h);
        int col = nt * 16 + (l & 15);
        bfb[i] = (_Float16)w1b[col * Cq + k];
    }
    __syncthreads();

    const int m16 = lane & 15, hi8 = (lane >= 16) ? 8 : 0;
    const int ntile  = (Bq * Nq) / 16;     // 1024
    const int nwaves = gridDim.x * W3;     // 512 -> exactly 2 tiles per wave

    for (int tile = blockIdx.x * W3 + wave; tile < ntile; tile += nwaves) {
        const int row0 = tile * 16;
        const h8* dro = (const h8*)(df + (size_t)(row0 + m16) * 192);
        v16h a[6];
        #pragma unroll
        for (int kc = 0; kc < 6; ++kc) {
            h8 lo = dro[4*kc + ((lane < 16) ? 0 : 1)];
            h8 hi = dro[4*kc + ((lane < 16) ? 2 : 3)];
            a[kc] = __builtin_shufflevector(lo, hi, 0,1,2,3,4,5,6,7,8,9,10,11,12,13,14,15);
        }
        #pragma unroll
        for (int nt = 0; nt < 4; ++nt) {
            v8f c = {};
            #pragma unroll
            for (int kc = 0; kc < 6; ++kc) {
                const v16h bf = *(const v16h*)&bfa[((kc * 4 + nt) * 32 + lane) * 16];
                c = __builtin_amdgcn_wmma_f32_16x16x32_f16(false, a[kc], false, bf,
                                                           (short)0, c, false, false);
            }
            const int col = nt * 16 + m16;
            const float s = s1[col], t = t1[col];
            #pragma unroll
            for (int r = 0; r < 8; ++r)
                h1s[wave][hi8 + r][col] = (_Float16)fmaxf(c[r] * s + t, 0.f);
        }
        __syncthreads();

        const h8* hro = (const h8*)&h1s[wave][m16][0];
        v16h a2[2];
        #pragma unroll
        for (int kc = 0; kc < 2; ++kc) {
            h8 lo = hro[4*kc + ((lane < 16) ? 0 : 1)];
            h8 hi = hro[4*kc + ((lane < 16) ? 2 : 3)];
            a2[kc] = __builtin_shufflevector(lo, hi, 0,1,2,3,4,5,6,7,8,9,10,11,12,13,14,15);
        }
        #pragma unroll
        for (int nt = 0; nt < 4; ++nt) {
            v8f c = {};
            #pragma unroll
            for (int kc = 0; kc < 2; ++kc) {
                const v16h bf = *(const v16h*)&bfb[((kc * 4 + nt) * 32 + lane) * 16];
                c = __builtin_amdgcn_wmma_f32_16x16x32_f16(false, a2[kc], false, bf,
                                                           (short)0, c, false, false);
            }
            const int col = nt * 16 + m16;
            #pragma unroll
            for (int r = 0; r < 8; ++r)
                out[(size_t)(row0 + hi8 + r) * Cq + col] = c[r] + bb1[col];
        }
        __syncthreads();
    }
}

// ---------------------------------------------------------------------------
extern "C" void kernel_launch(void* const* d_in, const int* in_sizes, int n_in,
                              void* d_out, int out_size, void* d_ws, size_t ws_size,
                              hipStream_t stream) {
    (void)in_sizes; (void)n_in; (void)out_size; (void)ws_size;
    const float* pcl     = (const float*)d_in[0];
    const float* noise   = (const float*)d_in[1];
    const float* feature = (const float*)d_in[2];
    const float* w2a  = (const float*)d_in[3];
    const float* b2a  = (const float*)d_in[4];
    const float* g2a  = (const float*)d_in[5];
    const float* bt2a = (const float*)d_in[6];
    const float* w2b  = (const float*)d_in[7];
    const float* b2b  = (const float*)d_in[8];
    const float* w1a  = (const float*)d_in[9];
    const float* b1a  = (const float*)d_in[10];
    const float* g1a  = (const float*)d_in[11];
    const float* bt1a = (const float*)d_in[12];
    const float* w1b  = (const float*)d_in[13];
    const float* b1b  = (const float*)d_in[14];
    float* out = (float*)d_out;

    char* ws = (char*)d_ws;
    int* close_idx = (int*)ws;                       //  64 KB
    int* knn       = close_idx + Bq * Nq;            // 960 KB (ends exactly at 1 MB)
    _Float16* df   = (_Float16*)(ws + (1 << 20));    //   6 MB, 16B aligned

    knn_kernel<<<(Bq * Nq) / 256, 256, 0, stream>>>(pcl, noise, close_idx, knn);
    point_mlp_kernel<<<256, 64, 0, stream>>>(pcl, noise, feature,
                                             w2a, b2a, g2a, bt2a, w2b, b2b,
                                             close_idx, knn, df);
    final_mlp_kernel<<<64, 256, 0, stream>>>(df, w1a, b1a, g1a, bt1a, w1b, b1b, out);
}
